// SparseBattery_55671366090953
// MI455X (gfx1250) — matrix operand order repaired and checked by reading, not compiled, not run on hardware
//
#include <hip/hip_runtime.h>

typedef __attribute__((ext_vector_type(16))) __bf16 v16bf;
typedef __attribute__((ext_vector_type(8)))  float  v8f;

union Frag16 {
  v16bf v;
  unsigned short u[16];
  uint4 q[2];
};

__device__ __forceinline__ unsigned short f2bf(float f) {
  unsigned int u = __float_as_uint(f);
  u += 0x7FFFu + ((u >> 16) & 1u);   // round-to-nearest-even
  return (unsigned short)(u >> 16);
}

// pack two floats into two bf16 (lo = a, hi = b)
__device__ __forceinline__ unsigned int f2bf2(float a, float b) {
  unsigned int ua = __float_as_uint(a);
  ua += 0x7FFFu + ((ua >> 16) & 1u);
  unsigned int ub = __float_as_uint(b);
  ub += 0x7FFFu + ((ub >> 16) & 1u);
  return (ua >> 16) | (ub & 0xFFFF0000u);
}

// ---------------- Kernel 1: pooled mean + sparsemax gate -----------------
__global__ void pool_gate_kernel(const float* __restrict__ x,
                                 const float* __restrict__ gate_w,
                                 const float* __restrict__ gate_b,
                                 float* __restrict__ g_out)
{
  __shared__ float pooled[512];
  __shared__ float zsh[8];
  const int b = blockIdx.x;
  const int tid = threadIdx.x;
  const int wv = tid >> 5;          // 8 waves (wave32)
  const int lane = tid & 31;
  const float* xb = x + (size_t)b * (512 * 784);

  // pooled[c] = mean over 784, one wave per channel round-robin
  for (int c = wv; c < 512; c += 8) {
    const float* row = xb + c * 784;
    float s = 0.f;
    for (int j = lane; j < 784; j += 32) s += row[j];
    #pragma unroll
    for (int o = 16; o >= 1; o >>= 1) s += __shfl_down(s, o, 32);
    if (lane == 0) pooled[c] = s * (1.0f / 784.0f);
  }
  __syncthreads();

  // z_k = pooled . gate_w[k] + gate_b[k], one wave per expert (K=8 waves)
  {
    float s = 0.f;
    const float* gw = gate_w + wv * 512;
    for (int c = lane; c < 512; c += 32) s += pooled[c] * gw[c];
    #pragma unroll
    for (int o = 16; o >= 1; o >>= 1) s += __shfl_down(s, o, 32);
    if (lane == 0) zsh[wv] = s + gate_b[wv];
  }
  __syncthreads();

  // sparsemax over K=8 (tiny, one thread)
  if (tid == 0) {
    float z[8], zs[8];
    #pragma unroll
    for (int k = 0; k < 8; ++k) { z[k] = zsh[k]; zs[k] = zsh[k]; }
    for (int a = 1; a < 8; ++a) {            // insertion sort, descending
      float v = zs[a]; int j = a - 1;
      while (j >= 0 && zs[j] < v) { zs[j + 1] = zs[j]; --j; }
      zs[j + 1] = v;
    }
    float cum = 0.f, cumk = 0.f; int kz = 1;
    #pragma unroll
    for (int a = 0; a < 8; ++a) {
      cum += zs[a];
      if (1.0f + (float)(a + 1) * zs[a] > cum) { kz = a + 1; cumk = cum; }
    }
    float tau = (cumk - 1.0f) / (float)kz;
    #pragma unroll
    for (int k = 0; k < 8; ++k) g_out[b * 8 + k] = fmaxf(z[k] - tau, 0.f);
  }
}

// ---------------- Kernel 2: fused w_eff construction + batched GEMM ------
// grid = (512/128 = 4 M-tiles, 64 batches), block = 256 threads (8 waves).
// A_lds holds the whole-K bf16 w_eff tile (128 x 512), fragment-packed so a
// wave's v_wmma_f32_16x16x32_bf16 A operand is two contiguous ds_load_b128.
__global__ __launch_bounds__(256)
void moe_gemm_kernel(const float* __restrict__ x,
                     const float* __restrict__ expert_w,
                     const float* __restrict__ g,
                     float* __restrict__ out)
{
  __shared__ unsigned short A_lds[128 * 512]; // 128 KB bf16, fragment-packed
  __shared__ unsigned short B_lds[7 * 512];   // 7 KB: 32x112 x-tile, packed
  __shared__ float gsh[8];

  const int tid  = threadIdx.x;
  const int wv   = tid >> 5;
  const int lane = tid & 31;
  const int lh   = lane >> 4;
  const int b    = blockIdx.y;
  const int m0   = blockIdx.x * 128;

  if (tid < 8) gsh[tid] = g[b * 8 + tid];
  __syncthreads();
  float gk[8];
  #pragma unroll
  for (int k = 0; k < 8; ++k) gk[k] = gsh[k];

  // ---- Stage A: w_eff[b][m0:m0+128][0:512] = sum_k g_k * expert_w[k] ----
  // Quad-vectorized: 4 consecutive Cin values share one contiguous pos run
  // in the fragment layout (i % 4 == 0 keeps lane-half and 8-group fixed),
  // so global reads are b128 (float4) and LDS writes are packed b64.
  {
    const float* ew = expert_w + (size_t)m0 * 512;
    #pragma unroll 2
    for (int e = 0; e < 64; ++e) {
      int idx = e * 256 + tid;        // 0..16383 quads
      int m  = idx >> 7;              // 0..127 (local Cout row)
      int i4 = (idx & 127) * 4;       // 0..508, Cin quad base (coalesced)
      const float* p = ew + m * 512 + i4;
      float s0 = 0.f, s1 = 0.f, s2 = 0.f, s3 = 0.f;
      #pragma unroll
      for (int k = 0; k < 8; ++k) {
        float4 w = *(const float4*)(p + k * 262144);
        s0 += gk[k] * w.x; s1 += gk[k] * w.y;
        s2 += gk[k] * w.z; s3 += gk[k] * w.w;
      }
      unsigned int lo = f2bf2(s0, s1);
      unsigned int hi = f2bf2(s2, s3);
      int kk  = i4 >> 5;
      int il  = i4 & 31;
      int ln  = ((il >> 3) & 1) * 16 + (m & 15);
      int pos = (il & 7) + ((il >> 4) << 3);        // multiple of 4
      unsigned int* dst =
          (unsigned int*)&A_lds[(((m >> 4) * 16) + kk) * 512 + ln * 16 + pos];
      dst[0] = lo;
      dst[1] = hi;
    }
  }

  const float* xb = x   + (size_t)b * (512 * 784);
  float*       ob = out + (size_t)b * (512 * 784);

  for (int nc = 0; nc < 7; ++nc) {          // N chunks of 112 (7 wmma tiles)
    const int n0 = nc * 112;
    v8f acc[7];
    #pragma unroll
    for (int nt = 0; nt < 7; ++nt) { v8f z = {0,0,0,0,0,0,0,0}; acc[nt] = z; }

    for (int ks = 0; ks < 16; ++ks) {       // K steps of 32 over Cin=512
      __syncthreads();                      // B_lds free (also covers A stage)
      // stage B: x[b][ks*32 .. +32][n0 .. n0+112] -> bf16, fragment-packed.
      // float2 global loads (8B aligned: row stride 784, n0 and c2 even).
      #pragma unroll
      for (int e = 0; e < 7; ++e) {         // 32*112/2 / 256 = 7
        int idx = e * 256 + tid;            // 0..1791 pairs
        int r  = idx / 56;                  // K row 0..31
        int c2 = (idx - r * 56) * 2;        // N col 0..110 even
        float2 v = *(const float2*)&xb[(ks * 32 + r) * 784 + n0 + c2];
        int ln   = (((r >> 3) & 1) * 16) + (c2 & 15);     // c2&15 <= 14
        int base = ((c2 >> 4) * 512) + (r & 7) + ((r >> 4) << 3);
        B_lds[base + ln * 16]        = f2bf(v.x);
        B_lds[base + (ln + 1) * 16]  = f2bf(v.y);
      }
      __syncthreads();

      Frag16 a;
      const uint4* ap = (const uint4*)&A_lds[(wv * 16 + ks) * 512 + lane * 16];
      a.q[0] = ap[0];
      a.q[1] = ap[1];

      #pragma unroll
      for (int nt = 0; nt < 7; ++nt) {
        Frag16 bf;
        const uint4* bp = (const uint4*)&B_lds[nt * 512 + lane * 16];
        bf.q[0] = bp[0];
        bf.q[1] = bp[1];
        acc[nt] = __builtin_amdgcn_wmma_f32_16x16x32_bf16(
            false, a.v, false, bf.v, (short)0, acc[nt], false, false);
      }
    }

    // epilogue: D layout -> VGPR r: M = r + 8*lh, N = lane&15
    const int rowb = m0 + wv * 16 + lh * 8;
    const int colb = n0 + (lane & 15);
    #pragma unroll
    for (int nt = 0; nt < 7; ++nt) {
      float* op = ob + rowb * 784 + colb + nt * 16;
      #pragma unroll
      for (int r = 0; r < 8; ++r) op[r * 784] = acc[nt][r];
    }
  }
}

extern "C" void kernel_launch(void* const* d_in, const int* in_sizes, int n_in,
                              void* d_out, int out_size, void* d_ws, size_t ws_size,
                              hipStream_t stream) {
  const float* x        = (const float*)d_in[0];   // (64,512,28,28)
  const float* gate_w   = (const float*)d_in[1];   // (8,512)
  const float* gate_b   = (const float*)d_in[2];   // (8,)
  const float* expert_w = (const float*)d_in[3];   // (8,512,512)
  float* out = (float*)d_out;                      // (64,512,28,28)
  float* gws = (float*)d_ws;                       // 64*8 gate weights

  pool_gate_kernel<<<dim3(64), dim3(256), 0, stream>>>(x, gate_w, gate_b, gws);
  moe_gemm_kernel<<<dim3(4, 64), dim3(256), 0, stream>>>(x, expert_w, gws, out);
}
